// VectorQuantizer_9036611190903
// MI455X (gfx1250) — matrix operand order repaired and checked by reading, not compiled, not run on hardware
//
#include <hip/hip_runtime.h>
#include <hip/hip_bf16.h>

typedef __attribute__((ext_vector_type(16))) _Float16 v16h;
typedef __attribute__((ext_vector_type(8)))  _Float16 v8h;
typedef __attribute__((ext_vector_type(4)))  _Float16 v4h;
typedef __attribute__((ext_vector_type(8)))  float    v8f;

#define KCB 512           // num_embeddings
#define DIM 512           // embedding dim
#define NPT 32768         // 32*32*32 latent vectors

static constexpr size_t QOFF  = 0;                          // quantized_st  [N, D]
static constexpr size_t LOFF  = (size_t)NPT * DIM;          // vq_loss scalar
static constexpr size_t OHOFF = LOFF + 1;                   // one_hot [N, K]
static constexpr size_t IOFF  = OHOFF + (size_t)NPT * KCB;  // indices [N] (as float)

// ---------------- prep: fp32 codebook -> f16 copy --------------------------
__global__ void vq_prep_convert(const float* __restrict__ W, _Float16* __restrict__ Wh) {
    int g = blockIdx.x * blockDim.x + threadIdx.x;          // 0..65535, 4 elems each
    float4 v = reinterpret_cast<const float4*>(W)[g];
    v4h h;
    h[0] = (_Float16)v.x; h[1] = (_Float16)v.y;
    h[2] = (_Float16)v.z; h[3] = (_Float16)v.w;
    reinterpret_cast<v4h*>(Wh)[g] = h;
}

// ---------------- prep: column norms (faithful dim=0 reduction) ------------
__global__ void vq_prep_colnorm(const float* __restrict__ W, float* __restrict__ colnorm) {
    int col = blockIdx.x * blockDim.x + threadIdx.x;
    if (col < DIM) {
        float s = 0.f;
        for (int i = 0; i < KCB; ++i) {
            float w = W[(size_t)i * DIM + col];
            s = fmaf(w, w, s);
        }
        colnorm[col] = s;
    }
}

// async memory->LDS copy of one 16B chunk (CDNA5 GLOBAL_LOAD_ASYNC_TO_LDS, GVS mode)
__device__ __forceinline__ void async_b128_to_lds(unsigned lds_addr, unsigned gbyte_off,
                                                  const _Float16* __restrict__ base) {
    asm volatile("global_load_async_to_lds_b128 %0, %1, %2"
                 :: "v"(lds_addr), "v"(gbyte_off), "s"(base)
                 : "memory");
}
__device__ __forceinline__ void wait_async0() {
    asm volatile("s_wait_asynccnt 0" ::: "memory");
}

// ---------------- main: WMMA distances + argmin + outputs ------------------
__global__ void __launch_bounds__(256)
vq_main(const float* __restrict__ latents, const float* __restrict__ W,
        const _Float16* __restrict__ Wh, const float* __restrict__ colnorm,
        float* __restrict__ out, float* __restrict__ partials) {
    // double-buffered codebook k-tile: 16 rows x 512 d (f16), rows padded by 8 halves
    __shared__ __align__(16) _Float16 shW[2][16 * 520];

    const int tid   = threadIdx.x;
    const int lane  = tid & 31;
    const int wv    = tid >> 5;           // wave in block (0..7)
    const int nloc  = lane & 15;
    const int hf    = lane >> 4;          // half of wave (0/1)
    const int waveGid = blockIdx.x * 8 + wv;
    const int n_base  = waveGid * 16;     // 16 latents per wave

    // this thread's 4 staging chunks (8 halves = 16B each), fixed across tiles
    int srow[4], scol[4];
    #pragma unroll
    for (int q = 0; q < 4; ++q) {
        int idx = tid + q * 256;          // 0..1023
        srow[q] = idx >> 6;
        scol[q] = (idx & 63) * 8;
    }

    // ---- load this wave's 16 latent rows as f16 B-fragments (once) ----
    // B (32x16, f16): lanes 0-15 hold K=0..7 & 16..23, lanes 16-31 hold K=8..15 & 24..31
    v16h bf[16];
    {
        const float* lrow = latents + (size_t)(n_base + nloc) * DIM;
        #pragma unroll
        for (int c = 0; c < 16; ++c) {
            const float4* p = (const float4*)(lrow + c * 32 + hf * 8);
            const float4* p2 = (const float4*)(lrow + c * 32 + 16 + hf * 8);
            float4 x0 = p[0], x1 = p[1], x2 = p2[0], x3 = p2[1];
            v16h b;
            b[0]  = (_Float16)x0.x; b[1]  = (_Float16)x0.y; b[2]  = (_Float16)x0.z; b[3]  = (_Float16)x0.w;
            b[4]  = (_Float16)x1.x; b[5]  = (_Float16)x1.y; b[6]  = (_Float16)x1.z; b[7]  = (_Float16)x1.w;
            b[8]  = (_Float16)x2.x; b[9]  = (_Float16)x2.y; b[10] = (_Float16)x2.z; b[11] = (_Float16)x2.w;
            b[12] = (_Float16)x3.x; b[13] = (_Float16)x3.y; b[14] = (_Float16)x3.z; b[15] = (_Float16)x3.w;
            bf[c] = b;
        }
    }

    // prologue: async-stage tile 0 into buffer 0
    #pragma unroll
    for (int q = 0; q < 4; ++q) {
        unsigned laddr = (unsigned)(size_t)&shW[0][srow[q] * 520 + scol[q]];
        unsigned goff  = (unsigned)(((0 * 16 + srow[q]) * DIM + scol[q]) * 2);
        async_b128_to_lds(laddr, goff, Wh);
    }
    wait_async0();
    __syncthreads();

    float minval = 3.402823466e38f;
    int   minidx = 0;

    for (int t = 0; t < 32; ++t) {
        const int buf = t & 1;
        // async-stage next tile into the other buffer while computing this one
        if (t < 31) {
            #pragma unroll
            for (int q = 0; q < 4; ++q) {
                unsigned laddr = (unsigned)(size_t)&shW[buf ^ 1][srow[q] * 520 + scol[q]];
                unsigned goff  = (unsigned)((((t + 1) * 16 + srow[q]) * DIM + scol[q]) * 2);
                async_b128_to_lds(laddr, goff, Wh);
            }
        }

        v8f acc = {0.f, 0.f, 0.f, 0.f, 0.f, 0.f, 0.f, 0.f};
        const _Float16* sw = &shW[buf][0];
        #pragma unroll
        for (int c = 0; c < 16; ++c) {
            // A (16x32, f16): lanes 0-15: K=0..7 & 16..23 ; lanes 16-31: K=8..15 & 24..31
            const v8h* a0p = (const v8h*)(&sw[nloc * 520 + c * 32 + hf * 8]);
            const v8h* a1p = (const v8h*)(&sw[nloc * 520 + c * 32 + 16 + hf * 8]);
            v8h a0 = *a0p, a1 = *a1p;
            v16h a = __builtin_shufflevector(a0, a1, 0,1,2,3,4,5,6,7,8,9,10,11,12,13,14,15);
            acc = __builtin_amdgcn_wmma_f32_16x16x32_f16(
                false, a, false, bf[c], (short)0, acc, false, false);
        }

        // dist[k][n] = colnorm[k] - 2*dot  (+||x||^2 constant per n, argmin-invariant)
        const float4* cp = (const float4*)(colnorm + t * 16 + hf * 8);
        float4 c0 = cp[0], c1 = cp[1];
        float cn[8] = {c0.x, c0.y, c0.z, c0.w, c1.x, c1.y, c1.z, c1.w};
        #pragma unroll
        for (int r = 0; r < 8; ++r) {
            float dv = fmaf(-2.f, acc[r], cn[r]);
            int kk = t * 16 + hf * 8 + r;       // D layout: VGPR r -> M = r + 8*half
            if (dv < minval) { minval = dv; minidx = kk; }
        }

        // my async writes for tile t+1 are done; barrier orders all waves'
        // buffer reads (dscnt drained by barrier) against next iter's writes
        wait_async0();
        __syncthreads();
    }

    // combine the two lane-halves (lane L holds column n = L&15)
    {
        float ov = __shfl_xor(minval, 16, 32);
        int   oi = __shfl_xor(minidx, 16, 32);
        if (ov < minval || (ov == minval && oi < minidx)) { minval = ov; minidx = oi; }
    }

    // ---- phase 2: quantized_st, one-hot, indices, loss partial ----
    float sum = 0.f;
    #pragma unroll 1
    for (int j = 0; j < 16; ++j) {
        int idxj = __shfl(minidx, j, 32);
        size_t n = (size_t)(n_base + j);
        const float* wrow = W + (size_t)idxj * DIM;
        const float* lr   = latents + n * DIM;
        float* qout  = out + QOFF  + n * DIM;
        float* ohout = out + OHOFF + n * (size_t)KCB;
        #pragma unroll
        for (int it = 0; it < 4; ++it) {
            int e = it * 128 + lane * 4;
            float4 w4 = *(const float4*)(wrow + e);
            float4 l4 = *(const float4*)(lr + e);
            float dx = w4.x - l4.x, dy = w4.y - l4.y;
            float dz = w4.z - l4.z, dw = w4.w - l4.w;
            sum += dx * dx + dy * dy + dz * dz + dw * dw;
            float4 q4 = { l4.x + dx, l4.y + dy, l4.z + dz, l4.w + dw }; // straight-through value
            *(float4*)(qout + e) = q4;
            float4 oh = { (e + 0 == idxj) ? 1.f : 0.f, (e + 1 == idxj) ? 1.f : 0.f,
                          (e + 2 == idxj) ? 1.f : 0.f, (e + 3 == idxj) ? 1.f : 0.f };
            *(float4*)(ohout + e) = oh;
        }
    }
    if (lane < 16) out[IOFF + n_base + lane] = (float)minidx;

    // wave-level loss partial (deterministic two-stage reduction)
    #pragma unroll
    for (int o = 16; o >= 1; o >>= 1) sum += __shfl_xor(sum, o, 32);
    if (lane == 0) partials[waveGid] = sum;
}

// ---------------- final deterministic loss reduction -----------------------
__global__ void vq_loss_reduce(const float* __restrict__ partials, float* __restrict__ out) {
    __shared__ float sm[256];
    float s = 0.f;
    for (int i = threadIdx.x; i < 2048; i += 256) s += partials[i];
    sm[threadIdx.x] = s;
    __syncthreads();
    for (int o = 128; o > 0; o >>= 1) {
        if (threadIdx.x < o) sm[threadIdx.x] += sm[threadIdx.x + o];
        __syncthreads();
    }
    // vq_loss = commitment*0.25 + embedding = 1.25 * mean((q - l)^2)
    if (threadIdx.x == 0) out[LOFF] = 1.25f * sm[0] / 16777216.0f;
}

extern "C" void kernel_launch(void* const* d_in, const int* in_sizes, int n_in,
                              void* d_out, int out_size, void* d_ws, size_t ws_size,
                              hipStream_t stream) {
    (void)in_sizes; (void)n_in; (void)out_size; (void)ws_size;
    const float* latents = (const float*)d_in[0];   // [32,32,32,512] fp32
    const float* W       = (const float*)d_in[1];   // [512,512] fp32
    float* out = (float*)d_out;

    char* ws = (char*)d_ws;
    _Float16* Wh      = (_Float16*)ws;                         // 512*512 f16 (512 KB)
    float*    colnorm = (float*)(ws + (size_t)KCB * DIM * 2);  // 512 f32
    float*    partials = colnorm + KCB;                        // 2048 f32

    vq_prep_convert<<<256, 256, 0, stream>>>(W, Wh);
    vq_prep_colnorm<<<2, 256, 0, stream>>>(W, colnorm);
    vq_main<<<256, 256, 0, stream>>>(latents, W, Wh, colnorm, out, partials);
    vq_loss_reduce<<<1, 256, 0, stream>>>(partials, out);
}
